// ZambaSSMLayer_69475390980167
// MI455X (gfx1250) — compile-verified, matrix operand order checked by reading, not run on hardware
//
#include <hip/hip_runtime.h>
#include <hip/hip_bf16.h>
#include <stdint.h>

// Problem constants (from reference)
#define D_MODEL 1024
#define D_STATE 16
#define D_CONV  4
#define D_INNER 2048
#define BB 4
#define TT 4096
#define E_TOTAL (D_INNER + 2 * D_STATE)   // 2080
#define M_TOTAL (BB * TT)                 // 16384

#define MT 256                            // GEMM M tile per workgroup
#define NT 32                             // GEMM N tile per workgroup

typedef __attribute__((ext_vector_type(16))) __bf16 v16bf;
typedef __attribute__((ext_vector_type(8)))  float  v8f;

// TDM path: this toolchain exposes the 6-arg builtin
// (uint32x4 g0, int32x8 g1, int32x4, int32x4, int32x8, i32 cpol)
#if defined(__AMDGCN__) && defined(__gfx1250__) && \
    __has_builtin(__builtin_amdgcn_tensor_load_to_lds) && \
    __has_builtin(__builtin_amdgcn_s_wait_tensorcnt)
#define USE_TDM 1
#else
#define USE_TDM 0
#endif

typedef unsigned int u32x4 __attribute__((ext_vector_type(4)));
typedef int          i32x8 __attribute__((ext_vector_type(8)));
typedef int          i32x4 __attribute__((ext_vector_type(4)));

// Issue one 2D TDM tile load: `rows` rows of 16 dwords (64B) each from a row-major
// bf16 matrix with row stride `strideDw` dwords, into LDS at byte offset ldsOff.
// D# fields packed per CDNA5 ISA.
__device__ __forceinline__ void tdm_load_2d(unsigned ldsOff, const uint16_t* gptr,
                                            unsigned rows, unsigned rowDw,
                                            unsigned strideDw) {
#if USE_TDM
    const unsigned long long ga = (unsigned long long)(uintptr_t)gptr;
    u32x4 g0;
    g0[0] = 1u;                                   // count=1, user mode, no gather
    g0[1] = ldsOff;                               // lds_addr (bytes)
    g0[2] = (unsigned)(ga & 0xffffffffu);         // global_addr[31:0]
    g0[3] = (unsigned)((ga >> 32) & 0x01ffffffu)  // global_addr[56:32]
          | 0x80000000u;                          // type=2 ("image"), bits 127:126
    i32x8 g1;
    g1[0] = (int)(2u << 16);                      // workgroup_mask=0, data_size=4B
    g1[1] = (int)((rowDw & 0xffffu) << 16);       // tensor_dim0[15:0]
    g1[2] = (int)(((rowDw >> 16) & 0xffffu)       // tensor_dim0[31:16]
          | ((rows & 0xffffu) << 16));            // tensor_dim1[15:0]
    g1[3] = (int)(((rows >> 16) & 0xffffu)        // tensor_dim1[31:16]
          | (16u << 16));                         // tile_dim0 = 16 dwords (64B row)
    g1[4] = (int)(rows & 0xffffu);                // tile_dim1 = rows, tile_dim2 = 0
    g1[5] = (int)strideDw;                        // tensor_dim0_stride[31:0]
    g1[6] = 0;                                    // stride hi / dim1_stride lo
    g1[7] = 0;
    const i32x4 z4 = {0, 0, 0, 0};                // groups 2/3 unused (2D tensor)
    const i32x8 z8 = {0, 0, 0, 0, 0, 0, 0, 0};
    __builtin_amdgcn_tensor_load_to_lds(g0, g1, z4, z4, z8, 0);
#else
    (void)ldsOff; (void)gptr; (void)rows; (void)rowDw; (void)strideDw;
#endif
}

// Round-to-nearest-even f32 -> bf16, packed pair (lo in [15:0], hi in [31:16])
__device__ __forceinline__ unsigned pack2_bf16(float lo, float hi) {
    unsigned a = __float_as_uint(lo);
    unsigned b = __float_as_uint(hi);
    a += 0x7fffu + ((a >> 16) & 1u);
    b += 0x7fffu + ((b >> 16) & 1u);
    return (a >> 16) | (b & 0xffff0000u);
}

// Bulk f32 -> bf16 conversion, 8 elements per thread.
__global__ __launch_bounds__(256)
void cvt_f32_bf16(const float* __restrict__ src, uint16_t* __restrict__ dst, size_t n8) {
    const size_t i = (size_t)blockIdx.x * 256 + threadIdx.x;
    if (i >= n8) return;
    const float4 a = *(const float4*)(src + i * 8);
    const float4 b = *(const float4*)(src + i * 8 + 4);
    uint4 p;
    p.x = pack2_bf16(a.x, a.y); p.y = pack2_bf16(a.z, a.w);
    p.z = pack2_bf16(b.x, b.y); p.w = pack2_bf16(b.z, b.w);
    *(uint4*)(dst + i * 8) = p;
}

// C[M x N] = A[M x K] * W[N x K]^T (+ res). A/W are bf16 row-major, K contiguous.
// Workgroup: 256 threads / 8 waves, tile MT(256) x NT(32); wave tile 32M x 32N
// (4 WMMA accumulators, A/B fragments each reused twice). K stepped by 32.
// Tiles land in double-buffered LDS via the Tensor Data Mover (wave 0 issues,
// TENSORcnt pipelined so DMA of step i+1 overlaps WMMA of step i); fallback is
// a cooperative bf16 copy when the TDM builtin is unavailable (e.g. host pass).
__global__ __launch_bounds__(256)
void gemm_bf16_wmma(const uint16_t* __restrict__ Abf, const uint16_t* __restrict__ Wbf,
                    const float* __restrict__ res, float* __restrict__ C,
                    int K, int ldC) {
    __shared__ __align__(16) unsigned aLds[2][MT * 16]; // 2 x 16 KB (bf16 tiles)
    __shared__ __align__(16) unsigned bLds[2][NT * 16]; // 2 x  2 KB

    const int tid  = threadIdx.x;
    const int wave = tid >> 5;
    const int lane = tid & 31;
    const int l16  = lane & 15;
    const int hiL  = (lane >= 16) ? 1 : 0;
    const int n0   = blockIdx.x * NT;
    const int m0   = blockIdx.y * MT;
    const int steps = K >> 5;
    const unsigned Kdw = (unsigned)(K >> 1);      // row length in dwords (bf16 pairs)

    v8f acc00 = {}, acc01 = {}, acc10 = {}, acc11 = {};

#if USE_TDM
    const unsigned aOff0 = (unsigned)(uintptr_t)&aLds[0][0];
    const unsigned aOff1 = (unsigned)(uintptr_t)&aLds[1][0];
    const unsigned bOff0 = (unsigned)(uintptr_t)&bLds[0][0];
    const unsigned bOff1 = (unsigned)(uintptr_t)&bLds[1][0];
    if (wave == 0) {  // prologue: DMA first K-slice into buffer 0
        tdm_load_2d(aOff0, Abf + (size_t)m0 * K, MT, Kdw, Kdw);
        tdm_load_2d(bOff0, Wbf + (size_t)n0 * K, NT, Kdw, Kdw);
    }
#endif

    for (int i = 0; i < steps; ++i) {
        const int buf = i & 1;
#if USE_TDM
        if (wave == 0) {
            if (i + 1 < steps) {
                const int k1 = (i + 1) << 5;
                tdm_load_2d(buf ? aOff0 : aOff1, Abf + (size_t)m0 * K + k1, MT, Kdw, Kdw);
                tdm_load_2d(buf ? bOff0 : bOff1, Wbf + (size_t)n0 * K + k1, NT, Kdw, Kdw);
                // 4 outstanding; TENSORcnt completes in-order per wave, so <=2
                // guarantees the pair feeding `buf` has landed in LDS.
                __builtin_amdgcn_s_wait_tensorcnt(2);
            } else {
                __builtin_amdgcn_s_wait_tensorcnt(0);
            }
        }
#else
        { // Cooperative bf16 tile copy (pure byte moves, no conversion).
            const int k0 = i << 5;
            #pragma unroll
            for (int j = 0; j < (MT * 16) / (256 * 4); ++j) {  // 1024 uint4, 4/thread
                const int idx = tid + j * 256;
                const int row = idx >> 2;         // 4 uint4 per 64B row
                const int q   = idx & 3;
                *(uint4*)&aLds[buf][row * 16 + q * 4] =
                    *(const uint4*)(Abf + (size_t)(m0 + row) * K + k0 + q * 8);
            }
            if (tid < (NT * 16) / 4) {            // 128 uint4
                const int row = tid >> 2;
                const int q   = tid & 3;
                *(uint4*)&bLds[buf][row * 16 + q * 4] =
                    *(const uint4*)(Wbf + (size_t)(n0 + row) * K + k0 + q * 8);
            }
        }
#endif
        __syncthreads();  // tile `buf` ready for all waves

        // Fragment loads per ISA 16-bit A/B layout:
        // lanes 0-15: K {0..7,16..23}; lanes 16-31: K {8..15,24..31}; 2 K / dword.
        union FragU { uint4 q[2]; v16bf v; } fa0, fa1, fb0, fb1;
        const int ks  = hiL ? 4 : 0;
        const int ar0 = (wave * 32 + l16) * 16;
        const int ar1 = ar0 + 16 * 16;
        fa0.q[0] = *(const uint4*)&aLds[buf][ar0 + ks];
        fa0.q[1] = *(const uint4*)&aLds[buf][ar0 + ks + 8];
        fa1.q[0] = *(const uint4*)&aLds[buf][ar1 + ks];
        fa1.q[1] = *(const uint4*)&aLds[buf][ar1 + ks + 8];
        const int br0 = l16 * 16;
        const int br1 = (16 + l16) * 16;
        fb0.q[0] = *(const uint4*)&bLds[buf][br0 + ks];
        fb0.q[1] = *(const uint4*)&bLds[buf][br0 + ks + 8];
        fb1.q[0] = *(const uint4*)&bLds[buf][br1 + ks];
        fb1.q[1] = *(const uint4*)&bLds[buf][br1 + ks + 8];

        acc00 = __builtin_amdgcn_wmma_f32_16x16x32_bf16(false, fa0.v, false, fb0.v,
                                                        (short)0, acc00, false, false);
        acc01 = __builtin_amdgcn_wmma_f32_16x16x32_bf16(false, fa0.v, false, fb1.v,
                                                        (short)0, acc01, false, false);
        acc10 = __builtin_amdgcn_wmma_f32_16x16x32_bf16(false, fa1.v, false, fb0.v,
                                                        (short)0, acc10, false, false);
        acc11 = __builtin_amdgcn_wmma_f32_16x16x32_bf16(false, fa1.v, false, fb1.v,
                                                        (short)0, acc11, false, false);

        __syncthreads();  // all waves done reading `buf` before it is overwritten
    }

    // C/D layout: VGPR r -> M = r (lanes 0-15) / r+8 (lanes 16-31), N = lane&15.
    const int mA = m0 + wave * 32 + (hiL ? 8 : 0);
    const int mB = mA + 16;
    const int nA = n0 + l16;
    const int nB = n0 + 16 + l16;
    #pragma unroll
    for (int r = 0; r < 8; ++r) {
        const size_t ra = (size_t)(mA + r) * ldC;
        const size_t rb = (size_t)(mB + r) * ldC;
        float v00 = acc00[r], v01 = acc01[r], v10 = acc10[r], v11 = acc11[r];
        if (res) {
            v00 += res[ra + nA]; v01 += res[ra + nB];
            v10 += res[rb + nA]; v11 += res[rb + nB];
        }
        C[ra + nA] = v00; C[ra + nB] = v01;
        C[rb + nA] = v10; C[rb + nB] = v11;
    }
}

// Depthwise causal conv (width 4) + bias + SiLU over the first D_INNER channels
// of proj. One thread per (b,t,d) element.
__global__ __launch_bounds__(256)
void conv_silu(const float* __restrict__ proj, const float* __restrict__ cw,
               const float* __restrict__ cb, float* __restrict__ u) {
    const size_t idx = (size_t)blockIdx.x * 256 + threadIdx.x;
    const int    d   = (int)(idx & (D_INNER - 1));
    const size_t bt  = idx >> 11;
    const int    t   = (int)(bt & (TT - 1));
    float acc = cb[d];
    #pragma unroll
    for (int j = 0; j < D_CONV; ++j) {
        const int tt = t - (D_CONV - 1) + j;
        if (tt >= 0)
            acc = fmaf(proj[(bt - (D_CONV - 1) + j) * E_TOTAL + d], cw[d * D_CONV + j], acc);
    }
    acc = acc / (1.0f + __expf(-acc));            // SiLU
    u[idx] = acc;
}

// Selective scan: one thread per (b,d) channel, 16-float state in registers.
// B_t/C_t (32 floats shared by the wave) fetched one-per-lane and broadcast
// with wave32 shuffles (ds_bpermute). y overwrites u in place.
__global__ __launch_bounds__(256)
void ssm_scan(const float* __restrict__ proj, const float* __restrict__ A_log,
              float* __restrict__ uy) {
    const int gid  = blockIdx.x * 256 + threadIdx.x;
    const int b    = gid >> 11;
    const int d    = gid & (D_INNER - 1);
    const int lane = threadIdx.x & 31;
    const float Ad = -__expf(A_log[d]);

    float h[D_STATE];
    #pragma unroll
    for (int s = 0; s < D_STATE; ++s) h[s] = 0.0f;

    const float* bcBase = proj + (size_t)b * TT * E_TOTAL + D_INNER + lane;
    float* uvp = uy + (size_t)b * TT * D_INNER + d;

    for (int t = 0; t < TT; ++t) {
        const float bc   = bcBase[(size_t)t * E_TOTAL];
        const float uval = uvp[(size_t)t * D_INNER];
        float y = 0.0f;
        #pragma unroll
        for (int s = 0; s < D_STATE; ++s) {
            const float bs = __shfl(bc, s, 32);
            const float cs = __shfl(bc, s + 16, 32);
            h[s] = fmaf(Ad, h[s], uval * bs);
            y    = fmaf(h[s], cs, y);
        }
        uvp[(size_t)t * D_INNER] = y;
    }
}

// In-place row LayerNorm over D_MODEL=1024. One 256-thread block per row.
__global__ __launch_bounds__(256)
void layernorm_inplace(float* __restrict__ out, const float* __restrict__ w,
                       const float* __restrict__ bb) {
    __shared__ float sdata[16];
    const int tid = threadIdx.x;
    float* p = out + (size_t)blockIdx.x * D_MODEL;

    float4 v = *(const float4*)(p + tid * 4);
    float s  = v.x + v.y + v.z + v.w;
    float ss = v.x * v.x + v.y * v.y + v.z * v.z + v.w * v.w;
    #pragma unroll
    for (int off = 16; off >= 1; off >>= 1) {
        s  += __shfl_down(s, off, 32);
        ss += __shfl_down(ss, off, 32);
    }
    const int wv_ = tid >> 5, lane = tid & 31;
    if (lane == 0) { sdata[wv_] = s; sdata[8 + wv_] = ss; }
    __syncthreads();
    if (tid == 0) {
        float S = 0.0f, SS = 0.0f;
        #pragma unroll
        for (int i = 0; i < 8; ++i) { S += sdata[i]; SS += sdata[8 + i]; }
        sdata[0] = S; sdata[8] = SS;
    }
    __syncthreads();
    const float mean = sdata[0] * (1.0f / D_MODEL);
    const float var  = sdata[8] * (1.0f / D_MODEL) - mean * mean;
    const float inv  = rsqrtf(var + 1e-5f);

    const float4 wv = *(const float4*)(w + tid * 4);
    const float4 bv = *(const float4*)(bb + tid * 4);
    float4 o;
    o.x = (v.x - mean) * inv * wv.x + bv.x;
    o.y = (v.y - mean) * inv * wv.y + bv.y;
    o.z = (v.z - mean) * inv * wv.z + bv.z;
    o.w = (v.w - mean) * inv * wv.w + bv.w;
    *(float4*)(p + tid * 4) = o;
}

extern "C" void kernel_launch(void* const* d_in, const int* in_sizes, int n_in,
                              void* d_out, int out_size, void* d_ws, size_t ws_size,
                              hipStream_t stream) {
    (void)in_sizes; (void)n_in; (void)out_size; (void)ws_size;
    const float* x          = (const float*)d_in[0];
    const float* in_proj_w  = (const float*)d_in[1];
    const float* conv_w     = (const float*)d_in[2];
    const float* conv_b     = (const float*)d_in[3];
    const float* A_log      = (const float*)d_in[4];
    const float* out_proj_w = (const float*)d_in[5];
    const float* ln_w       = (const float*)d_in[6];
    const float* ln_b       = (const float*)d_in[7];
    float* outp = (float*)d_out;

    // Workspace layout (bytes):
    //   proj  f32 [M x 2080]  (later reused for ybf)   136.3 MB
    //   uy    f32 [M x 2048]                           134.2 MB
    //   xbf   bf16[M x 1024]                            33.6 MB
    //   w1bf  bf16[2080 x 1024]                          4.3 MB
    //   w2bf  bf16[1024 x 2048]                          4.2 MB
    float*    proj = (float*)d_ws;
    float*    uy   = proj + (size_t)M_TOTAL * E_TOTAL;
    uint16_t* xbf  = (uint16_t*)(uy + (size_t)M_TOTAL * D_INNER);
    uint16_t* w1bf = xbf + (size_t)M_TOTAL * D_MODEL;
    uint16_t* w2bf = w1bf + (size_t)E_TOTAL * D_MODEL;
    uint16_t* ybf  = (uint16_t*)proj;   // proj is dead after the scan

    dim3 blk(256);

    // 0) f32 -> bf16 conversions (x and both weights are L2-resident afterwards)
    cvt_f32_bf16<<<dim3((unsigned)(((size_t)M_TOTAL * D_MODEL / 8) / 256)), blk, 0, stream>>>(
        x, xbf, (size_t)M_TOTAL * D_MODEL / 8);
    cvt_f32_bf16<<<dim3((unsigned)(((size_t)E_TOTAL * D_MODEL / 8 + 255) / 256)), blk, 0, stream>>>(
        in_proj_w, w1bf, (size_t)E_TOTAL * D_MODEL / 8);
    cvt_f32_bf16<<<dim3((unsigned)(((size_t)D_MODEL * D_INNER / 8) / 256)), blk, 0, stream>>>(
        out_proj_w, w2bf, (size_t)D_MODEL * D_INNER / 8);

    // 1) in_proj: proj = x @ in_proj_w^T   (M=16384, N=2080, K=1024)
    gemm_bf16_wmma<<<dim3(E_TOTAL / NT, M_TOTAL / MT), blk, 0, stream>>>(
        xbf, w1bf, nullptr, proj, D_MODEL, E_TOTAL);

    // 2) depthwise conv + SiLU -> u
    conv_silu<<<dim3((unsigned)((size_t)M_TOTAL * D_INNER / 256)), blk, 0, stream>>>(
        proj, conv_w, conv_b, uy);

    // 3) selective scan (y overwrites u in place)
    ssm_scan<<<dim3((BB * D_INNER) / 256), blk, 0, stream>>>(proj, A_log, uy);

    // 4) y -> bf16 (into the dead proj region)
    cvt_f32_bf16<<<dim3((unsigned)(((size_t)M_TOTAL * D_INNER / 8) / 256)), blk, 0, stream>>>(
        uy, ybf, (size_t)M_TOTAL * D_INNER / 8);

    // 5) out_proj + residual: out = y @ out_proj_w^T + x  (M=16384, N=1024, K=2048)
    gemm_bf16_wmma<<<dim3(D_MODEL / NT, M_TOTAL / MT), blk, 0, stream>>>(
        ybf, w2bf, x, outp, D_INNER, D_MODEL);

    // 6) LayerNorm in place
    layernorm_inplace<<<dim3(M_TOTAL), blk, 0, stream>>>(outp, ln_w, ln_b);
}
